// MultiHeadAttention_7730941133258
// MI455X (gfx1250) — matrix operand order repaired and checked by reading, not compiled, not run on hardware
//
#include <hip/hip_runtime.h>
#include <hip/hip_bf16.h>
#include <math.h>

// ---------------------------------------------------------------------------
// MI455X (gfx1250) multi-head attention, bf16 WMMA path + async-LDS staging.
//  B=2, S=2048, D=2048, H=16, hd=128.  wave32, V_WMMA_F32_16X16X32_BF16.
// ---------------------------------------------------------------------------

typedef __bf16 bf16_t;
typedef __attribute__((ext_vector_type(16))) __bf16 v16bf;
typedef __attribute__((ext_vector_type(8)))  __bf16 v8bf;
typedef __attribute__((ext_vector_type(8)))  float  v8f;
typedef __attribute__((ext_vector_type(4)))  int    v4i;

#if defined(__has_builtin)
#  if __has_builtin(__builtin_amdgcn_global_load_async_to_lds_b128)
#    define HAVE_ASYNC_COPY 1
#  endif
#endif

// 16B global -> LDS copy. Async (ASYNCcnt) when the gfx1250 builtin exists,
// else a synchronous fallback (DScnt-ordered, correct under __syncthreads()).
static __device__ __forceinline__ void copy16_g2l(const bf16_t* g, bf16_t* l) {
#ifdef HAVE_ASYNC_COPY
  __builtin_amdgcn_global_load_async_to_lds_b128((v4i*)g, (v4i*)l, 0, 0);
#else
  *(v8bf*)l = *(const v8bf*)g;
#endif
}

static __device__ __forceinline__ void async_wait_8() {
#ifdef HAVE_ASYNC_COPY
#  if __has_builtin(__builtin_amdgcn_s_wait_asynccnt)
  __builtin_amdgcn_s_wait_asynccnt(8);
#  else
  asm volatile("s_wait_asynccnt 0x8" ::: "memory");
#  endif
#endif
}
static __device__ __forceinline__ void async_wait_0() {
#ifdef HAVE_ASYNC_COPY
#  if __has_builtin(__builtin_amdgcn_s_wait_asynccnt)
  __builtin_amdgcn_s_wait_asynccnt(0);
#  else
  asm volatile("s_wait_asynccnt 0x0" ::: "memory");
#  endif
#endif
}

static __device__ __forceinline__ v16bf pack16(v8bf lo, v8bf hi) {
  v16bf r;
#pragma unroll
  for (int i = 0; i < 8; ++i) { r[i] = lo[i]; r[i + 8] = hi[i]; }
  return r;
}

// A fragment 16x32 (MxK), bf16, from row-major src (generic/global).
// ISA layout: lane(m=lane&15, hi=lane>>4): halfs[0..7]=K(hi*8..+8),
//             halfs[8..15]=K(16+hi*8..+8)  -> two contiguous 16B loads.
static __device__ __forceinline__ v16bf load_a(const bf16_t* p, int ld, int lane) {
  const int m = lane & 15, hi = lane >> 4;
  const bf16_t* b = p + (size_t)m * ld + hi * 8;
  return pack16(*(const v8bf*)b, *(const v8bf*)(b + 16));
}

// B fragment 32x16 (KxN) where B[k,n] = src[(n0+n)*ld + k0 + k] (src NxK).
// ISA layout: lane(n=lane&15, hi=lane>>4): halfs[0..7]=K(hi*8..+8),
//             halfs[8..15]=K(16+hi*8..+8).
static __device__ __forceinline__ v16bf load_b(const bf16_t* p, int ld, int lane) {
  const int n = lane & 15, hi = lane >> 4;
  const bf16_t* b = p + (size_t)n * ld + hi * 8;
  return pack16(*(const v8bf*)b, *(const v8bf*)(b + 16));
}

// Same fragment reads from an LDS tile with padded row stride of 40 elements
// (80B = 20 dword-banks -> conflict-spread for ds_read_b128).
static __device__ __forceinline__ v16bf lds_frag_a(const bf16_t* base, int lane) {
  const int m = lane & 15, hi = lane >> 4;
  const bf16_t* p = base + m * 40 + hi * 8;
  return pack16(*(const v8bf*)p, *(const v8bf*)(p + 16));
}
static __device__ __forceinline__ v16bf lds_frag_b(const bf16_t* base, int lane) {
  const int n = lane & 15, hi = lane >> 4;
  const bf16_t* p = base + n * 40 + hi * 8;
  return pack16(*(const v8bf*)p, *(const v8bf*)(p + 16));
}

static __device__ __forceinline__ v8f wmma_bf16(v16bf a, v16bf b, v8f c) {
  return __builtin_amdgcn_wmma_f32_16x16x32_bf16(false, a, false, b, (short)0, c,
                                                 false, false);
}

// ---------------------------------------------------------------------------
// fp32 -> bf16 elementwise convert
// ---------------------------------------------------------------------------
__global__ void f32_to_bf16(const float* __restrict__ in, bf16_t* __restrict__ out,
                            int n) {
  int i = blockIdx.x * blockDim.x + threadIdx.x;
  int stride = gridDim.x * blockDim.x;
  for (; i < n; i += stride) out[i] = (bf16_t)in[i];
}

// ---------------------------------------------------------------------------
// RoPE cos/sin tables: [S, 64] each.  inv_freq[d] = 10000^(-d/64)
// ---------------------------------------------------------------------------
__global__ void rope_tables(float* __restrict__ cosT, float* __restrict__ sinT) {
  const int s = blockIdx.x, d = threadIdx.x;  // 64 threads
  const float ln1e4 = 9.210340371976184f;     // ln(10000)
  float inv = __expf(-(float)d * (ln1e4 / 64.0f));
  float f = (float)s * inv;
  cosT[s * 64 + d] = cosf(f);
  sinT[s * 64 + d] = sinf(f);
}

// ---------------------------------------------------------------------------
// GEMM:  C[m,n] = sum_k A[m,k] * B[n,k]   (A @ B^T), bf16 in, f32 acc.
// Block: 128 threads = 4 waves (2x2), wave tile 64x64 (4x4 WMMA frags),
// block tile 128x128.  A/B tiles (128x32 each) double-buffered in LDS,
// filled with GLOBAL_LOAD_ASYNC_TO_LDS_B128 (ASYNCcnt, in-order per wave):
//   step t in flight + t+1 prefetched; s_wait_asynccnt 8 => step t landed.
// ---------------------------------------------------------------------------
template <bool OUT_F32>
__global__ __launch_bounds__(128) void gemm_xwT(
    const bf16_t* __restrict__ A, const bf16_t* __restrict__ B,
    float* __restrict__ Cf, bf16_t* __restrict__ Cb, int M, int N, int K) {
  __shared__ __align__(16) bf16_t Ash[2][128 * 40];
  __shared__ __align__(16) bf16_t Bsh[2][128 * 40];

  const int t = threadIdx.x;
  const int lane = t & 31, wv = t >> 5;
  const int m0b = blockIdx.y * 128, n0b = blockIdx.x * 128;
  const int mw = (wv >> 1) * 64, nw = (wv & 1) * 64;
  // Stage assignment: 4 threads per 64B row-chunk, 32 rows per pass, 4 passes.
  const int srow = t >> 2, sj = (t & 3) * 8;

  const v8f zero = {0.f, 0.f, 0.f, 0.f, 0.f, 0.f, 0.f, 0.f};
  v8f acc[4][4];
#pragma unroll
  for (int i = 0; i < 4; ++i)
#pragma unroll
    for (int j = 0; j < 4; ++j) acc[i][j] = zero;

  auto stage = [&](int k0, int buf) {
#pragma unroll
    for (int it = 0; it < 4; ++it) {
      const int r = it * 32 + srow;
      copy16_g2l(A + (size_t)(m0b + r) * K + k0 + sj, &Ash[buf][r * 40 + sj]);
      copy16_g2l(B + (size_t)(n0b + r) * K + k0 + sj, &Bsh[buf][r * 40 + sj]);
    }
  };

  const int NSTEP = K / 32;
  stage(0, 0);
  stage(32, 1);

  for (int s = 0; s < NSTEP; ++s) {
    if (s + 1 < NSTEP) async_wait_8(); else async_wait_0();
    __syncthreads();  // all waves' staged data visible
    const int buf = s & 1;
    v16bf a[4];
#pragma unroll
    for (int i = 0; i < 4; ++i)
      a[i] = lds_frag_a(&Ash[buf][(mw + 16 * i) * 40], lane);
#pragma unroll
    for (int j = 0; j < 4; ++j) {
      v16bf bf = lds_frag_b(&Bsh[buf][(nw + 16 * j) * 40], lane);
#pragma unroll
      for (int i = 0; i < 4; ++i) acc[i][j] = wmma_bf16(a[i], bf, acc[i][j]);
    }
    __syncthreads();  // everyone done reading buf before overwrite
    if (s + 2 < NSTEP) stage((s + 2) * 32, buf);
  }

  const int cn = lane & 15, ch = lane >> 4;
#pragma unroll
  for (int i = 0; i < 4; ++i)
#pragma unroll
    for (int j = 0; j < 4; ++j)
#pragma unroll
      for (int r = 0; r < 8; ++r) {
        const int row = m0b + mw + 16 * i + 8 * ch + r;
        const int col = n0b + nw + 16 * j + cn;
        if (OUT_F32)
          Cf[(size_t)row * N + col] = acc[i][j][r];
        else
          Cb[(size_t)row * N + col] = (bf16_t)acc[i][j][r];
      }
}

// ---------------------------------------------------------------------------
// RoPE + repack:  Qtmp/Ktmp/Vtmp [B,S,H*hd]  ->
//   Qr,Kr [B,H,S,hd] (rotated),  Vt [B,H,hd,S] (transposed for PV B-fragments)
// ---------------------------------------------------------------------------
__global__ __launch_bounds__(128) void rope_pack(
    const bf16_t* __restrict__ Q, const bf16_t* __restrict__ K,
    const bf16_t* __restrict__ V, const float* __restrict__ cosT,
    const float* __restrict__ sinT, bf16_t* __restrict__ Qr,
    bf16_t* __restrict__ Kr, bf16_t* __restrict__ Vt) {
  const int S = 2048, H = 16, HD = 128, D = 2048;
  const int s = blockIdx.x, h = blockIdx.y, b = blockIdx.z, d = threadIdx.x;
  const size_t src = ((size_t)b * S + s) * D + h * HD;
  const int dr = d & 63;
  const float c = cosT[s * 64 + dr], sn = sinT[s * 64 + dr];
  float q1 = (float)Q[src + d];
  float k1 = (float)K[src + d];
  if (d < 64) {
    const float q2 = (float)Q[src + d + 64], k2 = (float)K[src + d + 64];
    q1 = q1 * c - q2 * sn;
    k1 = k1 * c - k2 * sn;
  } else {
    const float q2 = (float)Q[src + d - 64], k2 = (float)K[src + d - 64];
    q1 = q1 * c + q2 * sn;
    k1 = k1 * c + k2 * sn;
  }
  const size_t bh = (size_t)b * H + h;
  const size_t dst = (bh * S + s) * HD + d;
  Qr[dst] = (bf16_t)q1;
  Kr[dst] = (bf16_t)k1;
  Vt[(bh * HD + d) * S + s] = V[src + d];
}

// ---------------------------------------------------------------------------
// Flash attention (non-causal).  Block = 64 q-rows, 4 waves, 16 rows/wave.
// k-tile = 64 keys.  Per wave per k-tile: 16 WMMA (QK^T) + 16 WMMA (PV).
// Online softmax: C-frag row = 8*hi + vgpr, so row stats reduce with
// __shfl_xor over the 16-lane half-wave (masks 1,2,4,8).
// P is staged through LDS (bf16) to convert C-layout -> A-fragment layout.
// ---------------------------------------------------------------------------
__global__ __launch_bounds__(128) void flash_attn(
    const bf16_t* __restrict__ Qr, const bf16_t* __restrict__ Kr,
    const bf16_t* __restrict__ Vt, bf16_t* __restrict__ O) {
  const int S = 2048, H = 16, HD = 128;
  const int qt = blockIdx.x, h = blockIdx.y, b = blockIdx.z;
  const int lane = threadIdx.x & 31, wv = threadIdx.x >> 5;
  const int hi = lane >> 4, nl = lane & 15;
  const size_t bh = (size_t)b * H + h;

  const bf16_t* Qb = Qr + (bh * S + (size_t)qt * 64 + wv * 16) * HD;
  const bf16_t* Kb = Kr + bh * S * HD;
  const bf16_t* Vb = Vt + bh * HD * S;

  __shared__ __align__(16) bf16_t Pl[4][16][72];  // 72 = 64 + pad

  v16bf qf[4];
#pragma unroll
  for (int c = 0; c < 4; ++c) qf[c] = load_a(Qb + c * 32, HD, lane);

  const v8f zero = {0.f, 0.f, 0.f, 0.f, 0.f, 0.f, 0.f, 0.f};
  v8f oacc[8];
#pragma unroll
  for (int i = 0; i < 8; ++i) oacc[i] = zero;
  float mrow[8], lrow[8];
#pragma unroll
  for (int r = 0; r < 8; ++r) { mrow[r] = -1e30f; lrow[r] = 0.f; }

  const float scale = 0.08838834764831845f;  // 1/sqrt(128)

  for (int k0 = 0; k0 < S; k0 += 64) {
    // ---- S = Q @ K^T (contract over hd) ----
    v8f sacc[4];
#pragma unroll
    for (int j = 0; j < 4; ++j) sacc[j] = zero;
#pragma unroll
    for (int j = 0; j < 4; ++j) {
#pragma unroll
      for (int c = 0; c < 4; ++c) {
        v16bf kb = load_b(Kb + (size_t)(k0 + j * 16) * HD + c * 32, HD, lane);
        sacc[j] = wmma_bf16(qf[c], kb, sacc[j]);
      }
    }
    // ---- online softmax update ----
#pragma unroll
    for (int r = 0; r < 8; ++r) {
      float sv[4];
      float mx = -1e30f;
#pragma unroll
      for (int j = 0; j < 4; ++j) {
        sv[j] = sacc[j][r] * scale;
        mx = fmaxf(mx, sv[j]);
      }
#pragma unroll
      for (int off = 1; off < 16; off <<= 1)
        mx = fmaxf(mx, __shfl_xor(mx, off, 32));
      const float mnew = fmaxf(mrow[r], mx);
      const float alpha = __expf(mrow[r] - mnew);
      float rsum = 0.f;
#pragma unroll
      for (int j = 0; j < 4; ++j) {
        const float p = __expf(sv[j] - mnew);
        sacc[j][r] = p;
        rsum += p;
      }
#pragma unroll
      for (int off = 1; off < 16; off <<= 1) rsum += __shfl_xor(rsum, off, 32);
      mrow[r] = mnew;
      lrow[r] = lrow[r] * alpha + rsum;
#pragma unroll
      for (int nf = 0; nf < 8; ++nf) oacc[nf][r] *= alpha;
    }
    // ---- stage P (C-layout) into LDS as bf16 ----
    __syncthreads();
#pragma unroll
    for (int j = 0; j < 4; ++j)
#pragma unroll
      for (int r = 0; r < 8; ++r)
        Pl[wv][r + 8 * hi][j * 16 + nl] = (bf16_t)sacc[j][r];
    __syncthreads();
    // ---- O += P @ V (contract over keys), B-frags from Vt (contiguous) ----
#pragma unroll
    for (int c2 = 0; c2 < 2; ++c2) {
      const bf16_t* prow = &Pl[wv][nl][c2 * 32 + hi * 8];
      v16bf pa = pack16(*(const v8bf*)prow, *(const v8bf*)(prow + 16));
#pragma unroll
      for (int nf = 0; nf < 8; ++nf) {
        v16bf vb = load_b(Vb + (size_t)(nf * 16) * S + k0 + c2 * 32, S, lane);
        oacc[nf] = wmma_bf16(pa, vb, oacc[nf]);
      }
    }
  }
  // ---- epilogue: normalize and write attn output [B,S,H*hd] as bf16 ----
#pragma unroll
  for (int nf = 0; nf < 8; ++nf)
#pragma unroll
    for (int r = 0; r < 8; ++r) {
      const int srow = qt * 64 + wv * 16 + 8 * hi + r;
      const float val = oacc[nf][r] / lrow[r];
      O[((size_t)b * S + srow) * 2048 + h * HD + nf * 16 + nl] = (bf16_t)val;
    }
}

// ---------------------------------------------------------------------------
// Host launch
// ---------------------------------------------------------------------------
extern "C" void kernel_launch(void* const* d_in, const int* in_sizes, int n_in,
                              void* d_out, int out_size, void* d_ws,
                              size_t ws_size, hipStream_t stream) {
  (void)in_sizes; (void)n_in; (void)out_size; (void)ws_size;
  const float* x  = (const float*)d_in[0];
  const float* Wq = (const float*)d_in[1];
  const float* Wk = (const float*)d_in[2];
  const float* Wv = (const float*)d_in[3];
  const float* Wo = (const float*)d_in[4];

  const int B = 2, S = 2048, D = 2048, H = 16;
  const size_t MS = (size_t)B * S;  // 4096 rows

  char* w = (char*)d_ws;
  size_t off = 0;
  auto alloc = [&](size_t bytes) -> char* {
    char* p = w + off;
    off += (bytes + 255) & ~((size_t)255);
    return p;
  };
  bf16_t* xb   = (bf16_t*)alloc(MS * D * 2);
  bf16_t* wqb  = (bf16_t*)alloc((size_t)D * D * 2);
  bf16_t* wkb  = (bf16_t*)alloc((size_t)D * D * 2);
  bf16_t* wvb  = (bf16_t*)alloc((size_t)D * D * 2);
  bf16_t* wob  = (bf16_t*)alloc((size_t)D * D * 2);
  bf16_t* qtmp = (bf16_t*)alloc(MS * D * 2);
  bf16_t* ktmp = (bf16_t*)alloc(MS * D * 2);
  bf16_t* vtmp = (bf16_t*)alloc(MS * D * 2);
  bf16_t* qr   = (bf16_t*)alloc(MS * D * 2);
  bf16_t* kr   = (bf16_t*)alloc(MS * D * 2);
  bf16_t* vt   = (bf16_t*)alloc(MS * D * 2);
  bf16_t* ao   = (bf16_t*)alloc(MS * D * 2);
  float* cosT  = (float*)alloc((size_t)S * 64 * 4);
  float* sinT  = (float*)alloc((size_t)S * 64 * 4);

  f32_to_bf16<<<1024, 256, 0, stream>>>(x, xb, (int)(MS * D));
  f32_to_bf16<<<1024, 256, 0, stream>>>(Wq, wqb, D * D);
  f32_to_bf16<<<1024, 256, 0, stream>>>(Wk, wkb, D * D);
  f32_to_bf16<<<1024, 256, 0, stream>>>(Wv, wvb, D * D);
  f32_to_bf16<<<1024, 256, 0, stream>>>(Wo, wob, D * D);
  rope_tables<<<S, 64, 0, stream>>>(cosT, sinT);

  dim3 gg(D / 128, (int)(MS / 128));  // (N tiles, M tiles) = (16, 32)
  gemm_xwT<false><<<gg, 128, 0, stream>>>(xb, wqb, nullptr, qtmp, (int)MS, D, D);
  gemm_xwT<false><<<gg, 128, 0, stream>>>(xb, wkb, nullptr, ktmp, (int)MS, D, D);
  gemm_xwT<false><<<gg, 128, 0, stream>>>(xb, wvb, nullptr, vtmp, (int)MS, D, D);

  rope_pack<<<dim3(S, H, B), 128, 0, stream>>>(qtmp, ktmp, vtmp, cosT, sinT, qr,
                                               kr, vt);
  flash_attn<<<dim3(S / 64, H, B), 128, 0, stream>>>(qr, kr, vt, ao);

  gemm_xwT<true><<<gg, 128, 0, stream>>>(ao, wob, (float*)d_out, nullptr,
                                         (int)MS, D, D);
}
